// MultiScaleAttention_53635551593219
// MI455X (gfx1250) — matrix operand order repaired and checked by reading, not compiled, not run on hardware
//
#include <hip/hip_runtime.h>
#include <hip/hip_bf16.h>

// ---------------------------------------------------------------------------
// MultiScaleAttention for MI455X (gfx1250), wave32 + v_wmma_f32_16x16x32_f16.
// Pipeline: f32->f16 convert -> fused QKV GEMM -> per-head flash windowed
// attention (S^T trick: per-lane softmax state, P stays in registers) ->
// output projection (+bias).
// ---------------------------------------------------------------------------

typedef __attribute__((ext_vector_type(16))) _Float16 h16;
typedef __attribute__((ext_vector_type(8)))  _Float16 h8v;
typedef __attribute__((ext_vector_type(8)))  float    f8v;

union H16 { h16 v; h8v h[2]; };

__device__ inline f8v wmma_f16(h16 a, h16 b, f8v c) {
  // D = A(16x32 f16) x B(32x16 f16) + C(16x16 f32)
  return __builtin_amdgcn_wmma_f32_16x16x32_f16(
      /*neg_a=*/false, a, /*neg_b=*/false, b,
      /*c_mod=*/(short)0, c, /*reuse_a=*/false, /*reuse_b=*/false);
}

// ---------------------------------------------------------------------------
__global__ void k_f32_to_f16(const float* __restrict__ in,
                             _Float16* __restrict__ out, int n) {
  int i = blockIdx.x * blockDim.x + threadIdx.x;
  if (i < n) out[i] = (_Float16)in[i];
}

// ---------------------------------------------------------------------------
// Fused QKV projection: X[12800x256] @ Wf^T[256x768] (Wf = [Wq;Wkv] rows).
// One wave per 16x16 output tile; scatter into q/k/v as [B, h, N, 32] f16.
__global__ __launch_bounds__(32) void k_qkv_gemm(
    const _Float16* __restrict__ X, const _Float16* __restrict__ Wf,
    _Float16* __restrict__ Q, _Float16* __restrict__ K,
    _Float16* __restrict__ V) {
  const int mtile = blockIdx.x, ntile = blockIdx.y;
  const int lane = threadIdx.x, lo = lane & 15, hi = lane >> 4;
  const int row = mtile * 16 + lo;   // token row (A-operand row for this lane)
  const int co  = ntile * 16 + lo;   // output channel (B-operand row of Wf)
  const _Float16* xr = X  + (size_t)row * 256;
  const _Float16* wr = Wf + (size_t)co * 256;
  f8v acc = {};
  for (int kk = 0; kk < 256; kk += 32) {
    H16 a, b;
    a.h[0] = *(const h8v*)(xr + kk + hi * 8);        // A: K = hi*8 .. +7
    a.h[1] = *(const h8v*)(xr + kk + 16 + hi * 8);   //    K = 16+hi*8 .. +7
    b.h[0] = *(const h8v*)(wr + kk + hi * 8);        // B: K = hi*8 .. +7
    b.h[1] = *(const h8v*)(wr + kk + 16 + hi * 8);   //    K = 16+hi*8 .. +7
    acc = wmma_f16(a.v, b.v, acc);
  }
  const int t    = co >> 8;          // 0=q, 1=k, 2=v
  const int head = (co & 255) >> 5;
  const int dim  = co & 31;
  _Float16* dst = (t == 0) ? Q : ((t == 1) ? K : V);
  for (int r = 0; r < 8; ++r) {
    int m = mtile * 16 + r + 8 * hi;           // D row = r + 8*hi
    int bidx = m / 6400, token = m - bidx * 6400;
    size_t idx = (((size_t)bidx * 8 + head) * 6400 + token) * 32 + dim;
    dst[idx] = (_Float16)acc[r];
  }
}

// ---------------------------------------------------------------------------
// Flash windowed attention for one head, 4 waves / block sharing K+V staging.
// Wave w handles query tile (blockIdx.x*4 + w) of window blockIdx.y.
// S^T = K x Q^T is computed so that (a) softmax state is one (m,l) scalar pair
// per lane (query = lane column), (b) P^T register layout IS the A-operand
// layout for the P*V WMMA -- no LDS round-trip for P.
// ps==1 -> global attention (w = N, one window), keys streamed in 32-chunks.
// Output flattened in WINDOW order (faithful to reference) into [B,N,C] f16.
__global__ __launch_bounds__(128) void k_attn(
    const _Float16* __restrict__ Q, const _Float16* __restrict__ K,
    const _Float16* __restrict__ V, _Float16* __restrict__ O,
    int head, int ps, int w, int nWp, int nqt, float scale) {
  const int N = 6400, Wimg = 80, C = 256;
  const int widx = blockIdx.y, b = blockIdx.z;
  const int tid = threadIdx.x;
  const int lane = tid & 31, lo = lane & 15, hi = (lane >> 4) & 1;
  const int wave = tid >> 5;

  __shared__ __align__(16) _Float16 kS[32][40];  // K chunk row-major [key][dim]
  __shared__ __align__(16) _Float16 vT[32][40];  // V chunk transposed [dim][key]

  auto tok = [&](int local) -> int {             // window-local -> token index
    if (ps == 1) return local;
    int pi = local / ps, pj = local - pi * ps;
    int wi = widx / nWp, wj = widx - wi * nWp;
    return (wi * ps + pi) * Wimg + (wj * ps + pj);
  };

  const size_t hb = ((size_t)b * 8 + head) * (size_t)N;

  // This wave's query tile (idle waves duplicate last tile, skip stores).
  int qt = blockIdx.x * 4 + wave;
  const bool active = (qt < nqt);
  if (qt >= nqt) qt = nqt - 1;

  // B-operand = Q^T (lane = query column lo, elements = dims hi*8..+7, 16+hi*8..+7).
  int ql = qt * 16 + lo;
  int qc = (ql < w) ? ql : (w - 1);
  const _Float16* qr = Q + (hb + tok(qc)) * 32;
  H16 bq;
  bq.h[0] = *(const h8v*)(qr + hi * 8);
  bq.h[1] = *(const h8v*)(qr + 16 + hi * 8);

  float mrun = -__builtin_inff(), lrun = 0.f;    // per-lane: query column lo
  f8v o0 = {}, o1 = {};                          // O rows = queries r+8*hi

  // Staging assignment: thread -> (key = tid&31, quarter-row = tid>>5).
  const int skey = tid & 31, sq = tid >> 5;

  const int nchunk = (w + 31) / 32;
  for (int ch = 0; ch < nchunk; ++ch) {
    const int base = ch * 32;

    // ---- cooperative staging: K row-major, V transposed -------------------
    {
      int kl = base + skey; if (kl >= w) kl = w - 1;
      size_t rbase = (hb + tok(kl)) * 32 + sq * 8;
      h8v kq = *(const h8v*)(K + rbase);
      h8v vq = *(const h8v*)(V + rbase);
      *(h8v*)&kS[skey][sq * 8] = kq;
      for (int e = 0; e < 8; ++e) vT[sq * 8 + e][skey] = vq[e];
      if (ch + 1 < nchunk) {                     // prefetch next chunk
        int nl = base + 32 + skey; if (nl >= w) nl = w - 1;
        __builtin_prefetch(K + (hb + tok(nl)) * 32 + sq * 8, 0, 1);
        __builtin_prefetch(V + (hb + tok(nl)) * 32 + sq * 8, 0, 1);
      }
    }
    __syncthreads();

    // ---- S^T tiles: A = K rows (keys lo / 16+lo), B = Q^T -----------------
    H16 ak0, ak1;
    ak0.h[0] = *(const h8v*)(&kS[lo][hi * 8]);
    ak0.h[1] = *(const h8v*)(&kS[lo][16 + hi * 8]);
    ak1.h[0] = *(const h8v*)(&kS[16 + lo][hi * 8]);
    ak1.h[1] = *(const h8v*)(&kS[16 + lo][16 + hi * 8]);
    f8v zero = {};
    f8v st0 = wmma_f16(ak0.v, bq.v, zero);   // rows = keys base+r+8*hi
    f8v st1 = wmma_f16(ak1.v, bq.v, zero);   // rows = keys base+16+r+8*hi

    // ---- online softmax, per-lane state (query = column lo) ---------------
    float v0[8], v1[8], cmax = -__builtin_inff();
    for (int r = 0; r < 8; ++r) {
      v0[r] = (base + r + 8 * hi < w)      ? st0[r] * scale : -__builtin_inff();
      v1[r] = (base + 16 + r + 8 * hi < w) ? st1[r] * scale : -__builtin_inff();
      cmax = fmaxf(cmax, fmaxf(v0[r], v1[r]));
    }
    cmax = fmaxf(cmax, __shfl_xor(cmax, 16));        // combine the two halves
    float mn = fmaxf(mrun, cmax);
    float sc = __expf(mrun - mn);
    mrun = mn;

    H16 ap;                                   // P^T regs == A-operand layout
    float rs = 0.f;
    for (int r = 0; r < 8; ++r) {
      float p0 = __expf(v0[r] - mn);          // key hi*8+r  -> A element r
      float p1 = __expf(v1[r] - mn);          // key 16+hi*8+r -> A element 8+r
      rs += p0 + p1;
      ap.h[0][r] = (_Float16)p0;
      ap.h[1][r] = (_Float16)p1;
    }
    rs += __shfl_xor(rs, 16);
    lrun = lrun * sc + rs;

    // Rescale O rows (row = query r+8*hi; its factor lives in lane r+8*hi).
    for (int r = 0; r < 8; ++r) {
      float s = __shfl(sc, r + 8 * hi);
      o0[r] *= s; o1[r] *= s;
    }

    // ---- P*V: B = V^T from LDS (lane = dim column lo / 16+lo) -------------
    H16 bv0, bv1;
    bv0.h[0] = *(const h8v*)(&vT[lo][hi * 8]);
    bv0.h[1] = *(const h8v*)(&vT[lo][16 + hi * 8]);
    bv1.h[0] = *(const h8v*)(&vT[16 + lo][hi * 8]);
    bv1.h[1] = *(const h8v*)(&vT[16 + lo][16 + hi * 8]);
    o0 = wmma_f16(ap.v, bv0.v, o0);
    o1 = wmma_f16(ap.v, bv1.v, o1);
    __syncthreads();
  }

  // ---- normalize + store in window-flattened order ------------------------
  float linv = 1.0f / lrun;
  for (int r = 0; r < 8; ++r) {
    int qi = qt * 16 + r + 8 * hi;
    if (!active || qi >= w) continue;
    float inv = __shfl(linv, r + 8 * hi);
    size_t ob = ((size_t)b * N + (size_t)widx * w + qi) * C + head * 32;
    O[ob + lo]      = (_Float16)(o0[r] * inv);
    O[ob + 16 + lo] = (_Float16)(o1[r] * inv);
  }
}

// ---------------------------------------------------------------------------
// Output projection: A[12800x256] @ Wproj^T + bias -> f32 d_out.
__global__ __launch_bounds__(32) void k_proj(
    const _Float16* __restrict__ A, const _Float16* __restrict__ Wp,
    const float* __restrict__ bias, float* __restrict__ out) {
  const int mtile = blockIdx.x, ntile = blockIdx.y;
  const int lane = threadIdx.x, lo = lane & 15, hi = lane >> 4;
  const int row = mtile * 16 + lo;
  const int co  = ntile * 16 + lo;
  const _Float16* ar = A  + (size_t)row * 256;
  const _Float16* wr = Wp + (size_t)co * 256;
  f8v acc = {};
  for (int kk = 0; kk < 256; kk += 32) {
    H16 a, b;
    a.h[0] = *(const h8v*)(ar + kk + hi * 8);
    a.h[1] = *(const h8v*)(ar + kk + 16 + hi * 8);
    b.h[0] = *(const h8v*)(wr + kk + hi * 8);
    b.h[1] = *(const h8v*)(wr + kk + 16 + hi * 8);
    acc = wmma_f16(a.v, b.v, acc);
  }
  const float bval = bias[co];
  for (int r = 0; r < 8; ++r) {
    int m = mtile * 16 + r + 8 * hi;
    out[(size_t)m * 256 + co] = acc[r] + bval;
  }
}

// ---------------------------------------------------------------------------
extern "C" void kernel_launch(void* const* d_in, const int* in_sizes, int n_in,
                              void* d_out, int out_size, void* d_ws,
                              size_t ws_size, hipStream_t stream) {
  (void)in_sizes; (void)n_in; (void)out_size; (void)ws_size;
  const float* x     = (const float*)d_in[0];   // [2,6400,256]
  const float* Wq    = (const float*)d_in[1];   // [256,256]
  const float* Wkv   = (const float*)d_in[2];   // [512,256]
  const float* Wproj = (const float*)d_in[3];   // [256,256]
  const float* bproj = (const float*)d_in[4];   // [256]

  char* ws = (char*)d_ws;
  size_t off = 0;
  auto take = [&](size_t nhalfs) {
    _Float16* p = (_Float16*)(ws + off);
    off = (off + nhalfs * 2 + 255) & ~(size_t)255;
    return p;
  };
  _Float16* xh  = take(12800u * 256);         // x as f16
  _Float16* wfh = take(768u * 256);           // [Wq;Wkv] as f16
  _Float16* wph = take(256u * 256);           // Wproj as f16
  _Float16* qh  = take(2u * 8 * 6400 * 32);   // q [B,h,N,d]
  _Float16* kh  = take(2u * 8 * 6400 * 32);   // k
  _Float16* vh  = take(2u * 8 * 6400 * 32);   // v
  _Float16* ah  = take(12800u * 256);         // attention out [B,N,C]

  k_f32_to_f16<<<(3276800 + 255) / 256, 256, 0, stream>>>(x, xh, 3276800);
  k_f32_to_f16<<<(65536 + 255) / 256, 256, 0, stream>>>(Wq, wfh, 65536);
  k_f32_to_f16<<<(131072 + 255) / 256, 256, 0, stream>>>(Wkv, wfh + 65536, 131072);
  k_f32_to_f16<<<(65536 + 255) / 256, 256, 0, stream>>>(Wproj, wph, 65536);

  k_qkv_gemm<<<dim3(800, 48), 32, 0, stream>>>(xh, wfh, qh, kh, vh);

  const int LOCAL[8] = {1, 2, 4, 5, 8, 10, 20, 40};
  const float scale = 0.17677669529663687f;  // 32^-0.5
  for (int h = 0; h < 8; ++h) {
    int ps = LOCAL[h], w, nW, nWp;
    if (ps == 1) { w = 6400; nW = 1; nWp = 1; }
    else { nWp = 80 / ps; nW = nWp * nWp; w = ps * ps; }
    int nqt = (w + 15) / 16;                 // query tiles per window
    dim3 grid((nqt + 3) / 4, nW, 2);         // 4 waves/block share K+V staging
    k_attn<<<grid, 128, 0, stream>>>(qh, kh, vh, ah, h, ps, w, nWp, nqt, scale);
  }

  k_proj<<<dim3(800, 16), 32, 0, stream>>>(ah, wph, bproj, (float*)d_out);
}
